// MultiResolutionModule_89146341196277
// MI455X (gfx1250) — matrix operand order repaired and checked by reading, not compiled, not run on hardware
//
#include <hip/hip_runtime.h>

// ---------------------------------------------------------------------------
// MultiResolutionModule for MI455X (gfx1250): bf16 WMMA implicit-GEMM convs,
// with async global->LDS weight staging (ASYNCcnt / s_wait_asynccnt).
// ---------------------------------------------------------------------------

typedef unsigned short u16;
typedef __attribute__((ext_vector_type(8)))  u16    u16x8;
typedef __attribute__((ext_vector_type(16))) u16    u16x16;
typedef __attribute__((ext_vector_type(16))) __bf16 v16bf;
typedef __attribute__((ext_vector_type(8)))  float  v8f;

union AV { u16x16 v; u16x8 h[2]; };

__device__ __forceinline__ u16 f32_to_bf16(float f) {
  union { float f; unsigned int u; } cv; cv.f = f;
  unsigned int u = cv.u;
  unsigned int r = u + 0x7FFFu + ((u >> 16) & 1u);   // round-to-nearest-even
  return (u16)(r >> 16);
}

// Stage `bytes` (multiple of 16) from global `src` into LDS at `ldsBase`,
// cooperatively across a 256-thread block, using CDNA5 async copies.
__device__ __forceinline__ void stage_weights_async(
    const u16* __restrict__ src, const u16* lds, unsigned bytes,
    unsigned tid) {
  unsigned ldsBase = (unsigned)(size_t)lds;   // low 32 bits == LDS offset
  for (unsigned off = tid * 16u; off < bytes; off += 256u * 16u) {
    unsigned l = ldsBase + off;
    asm volatile("global_load_async_to_lds_b128 %0, %1, %2"
                 :
                 : "v"(l), "v"(off), "s"(src)
                 : "memory");
  }
  asm volatile("s_wait_asynccnt 0x0" ::: "memory");
  __syncthreads();
}

__device__ __forceinline__ v16bf lds_load_b(const u16* p) {
  AV bv;
  bv.h[0] = *(const u16x8*)(p);
  bv.h[1] = *(const u16x8*)(p + 8);
  return __builtin_bit_cast(v16bf, bv.v);
}

// ---- problem dimensions ----------------------------------------------------
#define NB   2
#define D0_  32
#define H0_  128
#define W0_  128
#define C0_  32
#define D1_  16
#define H1_  64
#define W1_  64
#define C1_  64
#define N0_  (NB*D0_*H0_*W0_)      // 1,048,576 hi-res voxels
#define N1_  (NB*D1_*H1_*W1_)      // 131,072   lo-res voxels
#define E0_  (N0_*C0_)             // 33,554,432
#define E1_  (N1_*C1_)             // 8,388,608
#define EPS_ 1e-3f

// ---- stats buffer layout (floats) -----------------------------------------
#define S_CNT0  0
#define S_CNT1  1
#define S_CNT01 2
#define S_SUM00 16
#define S_SS00  48
#define S_SUM10 80
#define S_SS10  112
#define S_SUM01 144
#define S_SS01  208
#define S_SUM11 272
#define S_SS11  336
#define S_SC00  400
#define S_SH00  432
#define S_SC10  464
#define S_SH10  496
#define S_SC01  528
#define S_SH01  592
#define S_SC11  656
#define S_SH11  720
#define S_TOT   1024

// ===========================================================================
// Prep: mask-multiply activations -> bf16
// ===========================================================================
__global__ __launch_bounds__(256) void k_prep_x(
    const float* __restrict__ x, const int* __restrict__ m,
    u16* __restrict__ xm, int cshift, int total) {
  int i = blockIdx.x * blockDim.x + threadIdx.x;
  if (i >= total) return;
  float v = (m[i >> cshift] > 0) ? x[i] : 0.0f;
  xm[i] = f32_to_bf16(v);
}

// ===========================================================================
// Masks: int -> float, plus active-voxel count
// ===========================================================================
__global__ __launch_bounds__(256) void k_mask_count(
    const int* __restrict__ m, float* __restrict__ mf,
    float* __restrict__ cnt, int n) {
  __shared__ float red[256];
  int i = blockIdx.x * blockDim.x + threadIdx.x;
  float v = 0.0f;
  if (i < n) { v = (m[i] > 0) ? 1.0f : 0.0f; mf[i] = v; }
  red[threadIdx.x] = v;
  __syncthreads();
  for (int s = 128; s > 0; s >>= 1) {
    if (threadIdx.x < s) red[threadIdx.x] += red[threadIdx.x + s];
    __syncthreads();
  }
  if (threadIdx.x == 0) atomicAdd(cnt, red[0]);
}

// m01 = 3x3x3/s2/p1 max-pool of m0f, plus count
__global__ __launch_bounds__(256) void k_m01(
    const float* __restrict__ m0f, float* __restrict__ m01,
    float* __restrict__ cnt) {
  __shared__ float red[256];
  int v = blockIdx.x * blockDim.x + threadIdx.x;
  float mx = 0.0f;
  if (v < N1_) {
    int ow = v & (W1_ - 1);
    int oh = (v >> 6) & (H1_ - 1);
    int od = (v >> 12) & (D1_ - 1);
    int b  = v >> 16;
    for (int kd = 0; kd < 3; ++kd) {
      int id = 2 * od + kd - 1;
      if ((unsigned)id >= (unsigned)D0_) continue;
      for (int kh = 0; kh < 3; ++kh) {
        int ih = 2 * oh + kh - 1;
        if ((unsigned)ih >= (unsigned)H0_) continue;
        for (int kw = 0; kw < 3; ++kw) {
          int iw = 2 * ow + kw - 1;
          if ((unsigned)iw >= (unsigned)W0_) continue;
          float t = m0f[((size_t)((b * D0_ + id) * H0_ + ih)) * W0_ + iw];
          mx = fmaxf(mx, t);
        }
      }
    }
    m01[v] = mx;
  }
  red[threadIdx.x] = mx;
  __syncthreads();
  for (int s = 128; s > 0; s >>= 1) {
    if (threadIdx.x < s) red[threadIdx.x] += red[threadIdx.x + s];
    __syncthreads();
  }
  if (threadIdx.x == 0) atomicAdd(cnt, red[0]);
}

// ===========================================================================
// Weight swizzle: [t][ci][co] f32 -> per-lane WMMA B layout bf16
//   dst[((t*KS+s)*NT+n)*512 + L*16 + j] ; ci = s*32 + (j<8 ? 8*(L>>4)+j : 16+8*(L>>4)+(j-8)),
//   co = n*16 + (L&15)   (exact V_WMMA 32x16 bf16 B layout, wave32)
// ===========================================================================
__global__ __launch_bounds__(256) void k_swizzle_w(
    const float* __restrict__ src, u16* __restrict__ dst,
    int Cin, int Cout, int total) {
  int i = blockIdx.x * blockDim.x + threadIdx.x;
  if (i >= total) return;
  int NT = Cout >> 4;
  int KS = Cin >> 5;
  int j = i & 15;
  int L = (i >> 4) & 31;
  int n = (i >> 9) % NT;
  int s = (i / (512 * NT)) % KS;
  int t = i / (512 * NT * KS);
  int ci = s * 32 + ((j < 8) ? 8 * (L >> 4) + j : 16 + 8 * (L >> 4) + (j - 8));
  int co = n * 16 + (L & 15);
  dst[i] = f32_to_bf16(src[((size_t)(t * Cin + ci)) * Cout + co]);
}

// ===========================================================================
// Dense conv3d (k=3, pad=1) as implicit GEMM on v_wmma_f32_16x16x32_bf16.
// One wave: 16 output voxels (contig along W) x COUT channels.
// Whole swizzled weight set staged into LDS per block via async copies.
// ===========================================================================
template <int CIN, int COUT, int STRIDE>
__global__ __launch_bounds__(256) void k_conv_wmma(
    const u16* __restrict__ x,    // [NB,Di,Hi,Wi,CIN] bf16
    const u16* __restrict__ wsw,  // swizzled weights (global)
    float* __restrict__ y,        // [NB,Do,Ho,Wo,COUT] f32
    int Di, int Hi, int Wi, int Do, int Ho, int Wo) {
  constexpr int KS = CIN / 32;
  constexpr int NT = COUT / 16;
  constexpr unsigned WBYTES = 27u * KS * NT * 512u * 2u;
  extern __shared__ u16 smw[];

  // async global->LDS stage of all tap weights (shared by 8 waves)
  stage_weights_async(wsw, smw, WBYTES, threadIdx.x);

  int wave = blockIdx.x * (blockDim.x >> 5) + (threadIdx.x >> 5);
  int lane = threadIdx.x & 31;
  int tilesW = Wo >> 4;
  int total = NB * Do * Ho * tilesW;
  if (wave >= total) return;

  int tw = wave % tilesW;
  int t1 = wave / tilesW;
  int oh = t1 % Ho;
  int t2 = t1 / Ho;
  int od = t2 % Do;
  int b  = t2 / Do;
  int owBase = tw << 4;
  int mvox = lane & 15;
  int chunkBase = (lane & 16) ? 8 : 0;   // WMMA 16-bit A layout, wave32
  int ow = owBase + mvox;

  __builtin_prefetch(x + ((size_t)((b * Di + od * STRIDE) * Hi + oh * STRIDE)) * Wi * CIN,
                     0, 1);              // global_prefetch_b8

  v8f acc[NT];
#pragma unroll
  for (int n = 0; n < NT; ++n)
#pragma unroll
    for (int r = 0; r < 8; ++r) acc[n][r] = 0.0f;

  for (int kd = 0; kd < 3; ++kd) {
    int id = od * STRIDE + kd - 1;
    if ((unsigned)id >= (unsigned)Di) continue;
    for (int kh = 0; kh < 3; ++kh) {
      int ih = oh * STRIDE + kh - 1;
      if ((unsigned)ih >= (unsigned)Hi) continue;
      size_t rowBase = ((size_t)((b * Di + id) * Hi + ih)) * Wi;
      for (int kw = 0; kw < 3; ++kw) {
        int iw = ow * STRIDE + kw - 1;
        bool vz = (unsigned)iw < (unsigned)Wi;
        const u16* vox = x + (rowBase + iw) * CIN;
        int t = (kd * 3 + kh) * 3 + kw;
        const u16* wb = smw + (size_t)(t * KS) * NT * 512 + lane * 16;
#pragma unroll
        for (int s = 0; s < KS; ++s) {
          AV av;
          if (vz) {
            av.h[0] = *(const u16x8*)(vox + s * 32 + chunkBase);
            av.h[1] = *(const u16x8*)(vox + s * 32 + chunkBase + 16);
          } else {
#pragma unroll
            for (int e = 0; e < 16; ++e) av.v[e] = 0;
          }
          v16bf a = __builtin_bit_cast(v16bf, av.v);
#pragma unroll
          for (int n = 0; n < NT; ++n) {
            v16bf bm = lds_load_b(wb + (s * NT + n) * 512);
            acc[n] = __builtin_amdgcn_wmma_f32_16x16x32_bf16(
                false, a, false, bm, (short)0, acc[n], false, false);
          }
        }
      }
    }
  }

  // D layout: VGPR r -> M = r + 8*(lane>=16), N = lane%16
  int coBase = lane & 15;
  int mrow = (lane >> 4) << 3;
  size_t outRow = ((size_t)((b * Do + od) * Ho + oh)) * Wo;
#pragma unroll
  for (int n = 0; n < NT; ++n)
#pragma unroll
    for (int r = 0; r < 8; ++r)
      y[(outRow + owBase + mrow + r) * COUT + n * 16 + coBase] = acc[n][r];
}

// ===========================================================================
// Transposed conv (lhs_dilation=2, pad (1,2), k=3): C1 -> C0, doubles dims.
// out[o] += x[q/2] . w[k]   where q = o + k - 1, q even, q/2 in-range.
// ===========================================================================
__global__ __launch_bounds__(256) void k_invconv_wmma(
    const u16* __restrict__ x,    // [NB,D1,H1,W1,C1] bf16
    const u16* __restrict__ wsw,  // swizzled w10 (Cin=64 -> Cout=32)
    float* __restrict__ y) {      // [NB,D0,H0,W0,C0]
  constexpr int KS = C1_ / 32;   // 2
  constexpr int NT = C0_ / 16;   // 2
  constexpr unsigned WBYTES = 27u * KS * NT * 512u * 2u;
  extern __shared__ u16 smw[];

  stage_weights_async(wsw, smw, WBYTES, threadIdx.x);

  int wave = blockIdx.x * (blockDim.x >> 5) + (threadIdx.x >> 5);
  int lane = threadIdx.x & 31;
  int tilesW = W0_ >> 4;
  int total = NB * D0_ * H0_ * tilesW;
  if (wave >= total) return;

  int tw = wave % tilesW;
  int t1 = wave / tilesW;
  int oh = t1 % H0_;
  int t2 = t1 / H0_;
  int od = t2 % D0_;
  int b  = t2 / D0_;
  int owBase = tw << 4;
  int mvox = lane & 15;
  int chunkBase = (lane & 16) ? 8 : 0;
  int ow = owBase + mvox;

  v8f acc[NT];
#pragma unroll
  for (int n = 0; n < NT; ++n)
#pragma unroll
    for (int r = 0; r < 8; ++r) acc[n][r] = 0.0f;

  for (int kd = 0; kd < 3; ++kd) {
    int qd = od + kd - 1;
    if (qd & 1) continue;
    int id = qd >> 1;
    if ((unsigned)id >= (unsigned)D1_) continue;
    for (int kh = 0; kh < 3; ++kh) {
      int qh = oh + kh - 1;
      if (qh & 1) continue;
      int ih = qh >> 1;
      if ((unsigned)ih >= (unsigned)H1_) continue;
      size_t rowBase = ((size_t)((b * D1_ + id) * H1_ + ih)) * W1_;
      for (int kw = 0; kw < 3; ++kw) {
        int qw = ow + kw - 1;
        int iw = qw >> 1;
        bool vz = (!(qw & 1)) && ((unsigned)iw < (unsigned)W1_);
        const u16* vox = x + (rowBase + iw) * C1_;
        int t = (kd * 3 + kh) * 3 + kw;
        const u16* wb = smw + (size_t)(t * KS) * NT * 512 + lane * 16;
#pragma unroll
        for (int s = 0; s < KS; ++s) {
          AV av;
          if (vz) {
            av.h[0] = *(const u16x8*)(vox + s * 32 + chunkBase);
            av.h[1] = *(const u16x8*)(vox + s * 32 + chunkBase + 16);
          } else {
#pragma unroll
            for (int e = 0; e < 16; ++e) av.v[e] = 0;
          }
          v16bf a = __builtin_bit_cast(v16bf, av.v);
#pragma unroll
          for (int n = 0; n < NT; ++n) {
            v16bf bm = lds_load_b(wb + (s * NT + n) * 512);
            acc[n] = __builtin_amdgcn_wmma_f32_16x16x32_bf16(
                false, a, false, bm, (short)0, acc[n], false, false);
          }
        }
      }
    }
  }

  int coBase = lane & 15;
  int mrow = (lane >> 4) << 3;
  size_t outRow = ((size_t)((b * D0_ + od) * H0_ + oh)) * W0_;
#pragma unroll
  for (int n = 0; n < NT; ++n)
#pragma unroll
    for (int r = 0; r < 8; ++r)
      y[(outRow + owBase + mrow + r) * C0_ + n * 16 + coBase] = acc[n][r];
}

// ===========================================================================
// Masked per-channel sum / sum-of-squares reduction (for BN stats).
// blockDim.x must be a multiple of C (C in {32,64}).
// ===========================================================================
__global__ __launch_bounds__(256) void k_reduce_stats(
    const float* __restrict__ y, const float* __restrict__ mf,
    float* __restrict__ sumOut, float* __restrict__ ssOut,
    int C, int nvox) {
  __shared__ float s_sum[64];
  __shared__ float s_ss[64];
  if ((int)threadIdx.x < C) { s_sum[threadIdx.x] = 0.0f; s_ss[threadIdx.x] = 0.0f; }
  __syncthreads();
  int c = threadIdx.x % C;
  int vpi = blockDim.x / C;
  float lsum = 0.0f, lss = 0.0f;
  for (int v = blockIdx.x * vpi + (int)threadIdx.x / C; v < nvox;
       v += gridDim.x * vpi) {
    float val = y[(size_t)v * C + c] * mf[v];
    lsum += val;
    lss  += val * val;   // mask is {0,1}: (x*m)^2 == x^2*m
  }
  atomicAdd(&s_sum[c], lsum);
  atomicAdd(&s_ss[c], lss);
  __syncthreads();
  if ((int)threadIdx.x < C) {
    atomicAdd(&sumOut[c], s_sum[c]);
    atomicAdd(&ssOut[c], s_ss[c]);
  }
}

__global__ void k_init_stats(float* stats) {
  int i = blockIdx.x * blockDim.x + threadIdx.x;
  if (i < S_TOT) stats[i] = 0.0f;
}

// fold sums -> per-channel (scale, shift): y_norm = y*scale + shift
__global__ void k_finalize_stats(
    float* __restrict__ st,
    const float* __restrict__ g00, const float* __restrict__ b00,
    const float* __restrict__ g10, const float* __restrict__ b10,
    const float* __restrict__ g01, const float* __restrict__ b01,
    const float* __restrict__ g11, const float* __restrict__ b11) {
  int t = threadIdx.x;
  int c, sumI, ssI, scI, shI, cntI;
  const float *g, *bb;
  if (t < 32)       { c = t;       sumI = S_SUM00; ssI = S_SS00; scI = S_SC00; shI = S_SH00; cntI = S_CNT0;  g = g00; bb = b00; }
  else if (t < 64)  { c = t - 32;  sumI = S_SUM10; ssI = S_SS10; scI = S_SC10; shI = S_SH10; cntI = S_CNT0;  g = g10; bb = b10; }
  else if (t < 128) { c = t - 64;  sumI = S_SUM01; ssI = S_SS01; scI = S_SC01; shI = S_SH01; cntI = S_CNT01; g = g01; bb = b01; }
  else if (t < 192) { c = t - 128; sumI = S_SUM11; ssI = S_SS11; scI = S_SC11; shI = S_SH11; cntI = S_CNT1;  g = g11; bb = b11; }
  else return;
  float cnt  = fmaxf(st[cntI], 1.0f);
  float mean = st[sumI + c] / cnt;
  float var  = st[ssI + c] / cnt - mean * mean;
  float sc   = g[c] * rsqrtf(var + EPS_);
  st[scI + c] = sc;
  st[shI + c] = bb[c] - mean * sc;
}

// out0 = (relu(bn(y00)) + relu(bn(y10))) * m0f
__global__ __launch_bounds__(256) void k_final0(
    const float* __restrict__ y00, const float* __restrict__ y10,
    const float* __restrict__ m0f, const float* __restrict__ st,
    float* __restrict__ out) {
  int i = blockIdx.x * blockDim.x + threadIdx.x;
  if (i >= E0_) return;
  int c = i & 31;
  int v = i >> 5;
  float a = fmaxf(fmaf(y00[i], st[S_SC00 + c], st[S_SH00 + c]), 0.0f);
  float b = fmaxf(fmaf(y10[i], st[S_SC10 + c], st[S_SH10 + c]), 0.0f);
  out[i] = (a + b) * m0f[v];
}

// out1 = relu(bn(y01))*m01 + relu(bn(y11))*m1f
__global__ __launch_bounds__(256) void k_final1(
    const float* __restrict__ y01, const float* __restrict__ y11,
    const float* __restrict__ m01, const float* __restrict__ m1f,
    const float* __restrict__ st, float* __restrict__ out) {
  int i = blockIdx.x * blockDim.x + threadIdx.x;
  if (i >= E1_) return;
  int c = i & 63;
  int v = i >> 6;
  float a = fmaxf(fmaf(y01[i], st[S_SC01 + c], st[S_SH01 + c]), 0.0f) * m01[v];
  float b = fmaxf(fmaf(y11[i], st[S_SC11 + c], st[S_SH11 + c]), 0.0f) * m1f[v];
  out[i] = a + b;
}

// ===========================================================================
extern "C" void kernel_launch(void* const* d_in, const int* in_sizes, int n_in,
                              void* d_out, int out_size, void* d_ws,
                              size_t ws_size, hipStream_t stream) {
  (void)in_sizes; (void)n_in; (void)out_size; (void)ws_size;

  const float* x0  = (const float*)d_in[0];
  const float* x1  = (const float*)d_in[1];
  const int*   m0  = (const int*)d_in[2];
  const int*   m1  = (const int*)d_in[3];
  const float* w00 = (const float*)d_in[4];
  const float* g00 = (const float*)d_in[5];
  const float* b00 = (const float*)d_in[6];
  const float* w10 = (const float*)d_in[7];
  const float* g10 = (const float*)d_in[8];
  const float* b10 = (const float*)d_in[9];
  const float* w01 = (const float*)d_in[10];
  const float* g01 = (const float*)d_in[11];
  const float* b01 = (const float*)d_in[12];
  const float* w11 = (const float*)d_in[13];
  const float* g11 = (const float*)d_in[14];
  const float* b11 = (const float*)d_in[15];

  char* ws = (char*)d_ws;
  const size_t OFF_X0M  = 0;
  const size_t OFF_X1M  = OFF_X0M + (size_t)E0_ * 2;
  const size_t OFF_Y00  = OFF_X1M + (size_t)E1_ * 2;
  const size_t OFF_Y10  = OFF_Y00 + (size_t)E0_ * 4;
  const size_t OFF_Y01  = OFF_Y10 + (size_t)E0_ * 4;
  const size_t OFF_Y11  = OFF_Y01 + (size_t)E1_ * 4;
  const size_t OFF_M0F  = OFF_Y11 + (size_t)E1_ * 4;
  const size_t OFF_M1F  = OFF_M0F + (size_t)N0_ * 4;
  const size_t OFF_M01  = OFF_M1F + (size_t)N1_ * 4;
  const size_t OFF_W00S = OFF_M01 + (size_t)N1_ * 4;
  const size_t OFF_W10S = OFF_W00S + (size_t)27648 * 2;
  const size_t OFF_W01S = OFF_W10S + (size_t)55296 * 2;
  const size_t OFF_W11S = OFF_W01S + (size_t)55296 * 2;
  const size_t OFF_STAT = OFF_W11S + (size_t)110592 * 2;

  u16*   x0m  = (u16*)(ws + OFF_X0M);
  u16*   x1m  = (u16*)(ws + OFF_X1M);
  float* y00  = (float*)(ws + OFF_Y00);
  float* y10  = (float*)(ws + OFF_Y10);
  float* y01  = (float*)(ws + OFF_Y01);
  float* y11  = (float*)(ws + OFF_Y11);
  float* m0f  = (float*)(ws + OFF_M0F);
  float* m1f  = (float*)(ws + OFF_M1F);
  float* m01v = (float*)(ws + OFF_M01);
  u16*   w00s = (u16*)(ws + OFF_W00S);
  u16*   w10s = (u16*)(ws + OFF_W10S);
  u16*   w01s = (u16*)(ws + OFF_W01S);
  u16*   w11s = (u16*)(ws + OFF_W11S);
  float* st   = (float*)(ws + OFF_STAT);

  float* out0 = (float*)d_out;
  float* out1 = out0 + E0_;

  // 0) stats = 0
  k_init_stats<<<4, 256, 0, stream>>>(st);

  // 1) masked bf16 activations
  k_prep_x<<<(E0_ + 255) / 256, 256, 0, stream>>>(x0, m0, x0m, 5, E0_);
  k_prep_x<<<(E1_ + 255) / 256, 256, 0, stream>>>(x1, m1, x1m, 6, E1_);

  // 2) float masks + counts, strided-conv output mask
  k_mask_count<<<(N0_ + 255) / 256, 256, 0, stream>>>(m0, m0f, st + S_CNT0, N0_);
  k_mask_count<<<(N1_ + 255) / 256, 256, 0, stream>>>(m1, m1f, st + S_CNT1, N1_);
  k_m01<<<(N1_ + 255) / 256, 256, 0, stream>>>(m0f, m01v, st + S_CNT01);

  // 3) weight swizzle into WMMA B-layout
  k_swizzle_w<<<(27648 + 255) / 256, 256, 0, stream>>>(w00, w00s, 32, 32, 27648);
  k_swizzle_w<<<(55296 + 255) / 256, 256, 0, stream>>>(w10, w10s, 64, 32, 55296);
  k_swizzle_w<<<(55296 + 255) / 256, 256, 0, stream>>>(w01, w01s, 32, 64, 55296);
  k_swizzle_w<<<(110592 + 255) / 256, 256, 0, stream>>>(w11, w11s, 64, 64, 110592);

  // 4) the four convolutions as WMMA implicit GEMMs (8 waves / block),
  //    each block async-stages its swizzled weights into LDS.
  {
    int waves = NB * D0_ * H0_ * (W0_ / 16);               // 65536
    k_conv_wmma<32, 32, 1><<<(waves + 7) / 8, 256, 55296, stream>>>(
        x0m, w00s, y00, D0_, H0_, W0_, D0_, H0_, W0_);
    k_invconv_wmma<<<(waves + 7) / 8, 256, 110592, stream>>>(x1m, w10s, y10);
  }
  {
    int waves = NB * D1_ * H1_ * (W1_ / 16);               // 8192
    k_conv_wmma<32, 64, 2><<<(waves + 7) / 8, 256, 110592, stream>>>(
        x0m, w01s, y01, D0_, H0_, W0_, D1_, H1_, W1_);
    k_conv_wmma<64, 64, 1><<<(waves + 7) / 8, 256, 221184, stream>>>(
        x1m, w11s, y11, D1_, H1_, W1_, D1_, H1_, W1_);
  }

  // 5) masked BN statistics
  k_reduce_stats<<<1024, 256, 0, stream>>>(y00, m0f, st + S_SUM00, st + S_SS00, 32, N0_);
  k_reduce_stats<<<1024, 256, 0, stream>>>(y10, m0f, st + S_SUM10, st + S_SS10, 32, N0_);
  k_reduce_stats<<<1024, 256, 0, stream>>>(y01, m01v, st + S_SUM01, st + S_SS01, 64, N1_);
  k_reduce_stats<<<1024, 256, 0, stream>>>(y11, m1f, st + S_SUM11, st + S_SS11, 64, N1_);
  k_finalize_stats<<<1, 192, 0, stream>>>(st, g00, b00, g10, b10, g01, b01, g11, b11);

  // 6) fused BN + ReLU + mask + branch-sum
  k_final0<<<(E0_ + 255) / 256, 256, 0, stream>>>(y00, y10, m0f, st, out0);
  k_final1<<<(E1_ + 255) / 256, 256, 0, stream>>>(y01, y11, m01v, m1f, st, out1);
}
